// SpatialLanguageModel_53876069761174
// MI455X (gfx1250) — matrix operand order repaired and checked by reading, not compiled, not run on hardware
//
#include <hip/hip_runtime.h>
#include <math.h>

// ---------------- problem constants ----------------
constexpr int  kV  = 50257;
constexpr int  kT  = 1024;
constexpr int  kD  = 1024;
constexpr int  kH  = 16;
constexpr int  kL  = 4;
constexpr int  kDS = 3;
constexpr int  kB  = 2;
constexpr int  kDH = kD / kH;     // 64
constexpr int  kDF = 4 * kD;      // 4096
constexpr float kPEN = 1.0f, kEW = 0.01f, kFW = 0.1f, kRW = 0.01f;

constexpr long long LOGITS_N = (long long)kB * kT * kV;            // 102,926,336
constexpr long long W_N      = (long long)kB * kH * kT * kT;       // 33,554,432 per layer
constexpr long long D_N      = (long long)kT * kT;                 // 1,048,576 per layer
constexpr long long SCALAR_OFF = LOGITS_N;                         // loss,tde,tfe,tre
constexpr long long W_OFF    = LOGITS_N + 4;
constexpr long long DT_OFF   = W_OFF + (long long)kL * W_N;

// workspace layout (in floats)
constexpr long long WS_ACC  = 0;                     // 4 accumulators: tde,tfe,tre,ce
constexpr long long WS_X    = 64;                                  // [B,T,D]
constexpr long long WS_H    = WS_X   + (long long)kB * kT * kD;    // [B,T,D]
constexpr long long WS_QKV  = WS_H   + (long long)kB * kT * kD;    // [B,T,3D]
constexpr long long WS_ATT  = WS_QKV + (long long)kB * kT * 3 * kD;// [B,T,D]
constexpr long long WS_MLP  = WS_ATT + (long long)kB * kT * kD;    // [B,T,DF]
constexpr long long WS_EMBBF= WS_MLP + (long long)kB * kT * kDF;   // [V,D] bf16 (as half-floats)

// ---------------- WMMA / TDM types ----------------
typedef __bf16 bf16_t;
typedef __attribute__((ext_vector_type(16))) bf16_t v16bf;
typedef __attribute__((ext_vector_type(8)))  float  v8f;
typedef __attribute__((ext_vector_type(4)))  unsigned int u32x4;
typedef __attribute__((ext_vector_type(8)))  int    i32x8;
typedef __attribute__((ext_vector_type(4)))  int    i32x4;

union Frag { v16bf v; unsigned int u[8]; };

#if defined(__has_builtin)
#  if __has_builtin(__builtin_amdgcn_tensor_load_to_lds) && __has_builtin(__builtin_amdgcn_s_wait_tensorcnt)
#    define HAS_TDM 1
#  endif
#endif
#ifndef HAS_TDM
#  define HAS_TDM 0
#endif

// Native bf16 convert (RNE) packed into one dword -> lets clang pick v_cvt ops.
__device__ __forceinline__ unsigned int pack2(float lo, float hi) {
  union { bf16_t h[2]; unsigned int u; } r;
  r.h[0] = (bf16_t)lo;
  r.h[1] = (bf16_t)hi;
  return r.u;
}

#if HAS_TDM
// TDM: DMA a 2D bf16 tile [rows=tileN x cols=32] from global into LDS with padded
// row pitch of 18 DWORDs ([n][kpair] layout). OOB rows (>= dimN) return zero.
// D# per CDNA5 ISA (group0 128b, group1 256b; groups 2/3 unused).
// 6-arg builtin form (clang-23 / amdgpu-toolchain lane).
__device__ __forceinline__ void tdm_load_tile_bf16(
    const unsigned short* gbase, long long elemOff, int pitchElems,
    int dimN, unsigned ldsByteOff)
{
  unsigned long long ga = (unsigned long long)(uintptr_t)gbase + 2ull * (unsigned long long)elemOff;
  u32x4 g0;
  g0[0] = 1u;                                        // count=1, user mode
  g0[1] = ldsByteOff;                                // lds_addr [63:32]
  g0[2] = (unsigned)(ga & 0xFFFFFFFFull);            // global_addr [95:64]
  g0[3] = (unsigned)((ga >> 32) & 0x1FFFFFFull) | (2u << 30);  // addr[56:32] | type=2
  unsigned td0 = 0x7FFFFFFFu;                        // dim0 never OOB (K mult of 32)
  unsigned td1 = (unsigned)dimN;                     // rows remaining -> zero-fill tail
  i32x8 g1;
  g1[0] = (int)((1u << 16) | (1u << 20) | (3u << 22) | (1u << 25));
          // data_size=2B, pad_enable, pad_interval=16DW, pad_amount=2DW
  g1[1] = (int)((td0 & 0xFFFFu) << 16);              // tensor_dim0 low
  g1[2] = (int)(((td0 >> 16) & 0xFFFFu) | ((td1 & 0xFFFFu) << 16));
  g1[3] = (int)(((td1 >> 16) & 0xFFFFu) | (32u << 16));   // tile_dim0 = 32
  g1[4] = 128;                                       // tile_dim1 = 128 (tile_dim2=0)
  g1[5] = pitchElems;                                // tensor_dim0_stride low 32
  g1[6] = 0;
  g1[7] = 0;
  i32x4 gz4 = {0, 0, 0, 0};
  i32x8 gz8 = {0, 0, 0, 0, 0, 0, 0, 0};
  __builtin_amdgcn_tensor_load_to_lds(g0, g1, gz4, gz4, gz8, 0);
}
#endif

// ---------------- generic bf16-WMMA GEMM ----------------
// Block tile: BM=64 x BN (64 or 128), BK=32, 256 threads = 8 waves.
// Double-buffered LDS: stage tile k+1 while 4 WMMAs run on tile k (1 barrier/iter).
// Wave (w): m-tile = w&3 (16 rows), n-tiles = (w>>2)*TPW .. +TPW-1 (TPW = BN/32).
// TRANSB: global B is [N,K] row-major (k / emb) -> direct copy into packed [n][kpair] LDS.
// SCORE : val = val*scale - PEN*dists[m*T+n]; causal mask n>m -> -inf.
// GUARD : bounds-check C stores on n (lm-head ragged N).
// TDMB  : B tiles DMA'd by the Tensor Data Mover from a prepacked bf16 matrix Bbf.
template<int BN, bool TRANSB, bool BIAS, bool GELU, bool RES, bool SCORE, bool GUARD, bool TDMB>
__global__ __launch_bounds__(256) void gemm_wmma(
    const float* __restrict__ A, const float* __restrict__ Bm,
    const unsigned short* __restrict__ Bbf,
    const float* __restrict__ bias, const float* __restrict__ resid,
    float* __restrict__ C, const float* __restrict__ dists,
    int M, int N, int K, int lda, int ldb, int ldc,
    long long aOuter, long long aInner, long long bOuter, long long bInner,
    long long cOuter, long long cInner, int zdiv, float scale)
{
  constexpr int TPW = BN / 32;                 // n-tiles per wave (2 or 4)
  constexpr int ASTR = 18;                     // A LDS row stride (uints), 8B-aligned pairs
  constexpr int BR = TRANSB ? BN : 16;         // B LDS rows
  constexpr int BC = TRANSB ? 18 : BN + 4;     // B LDS cols (uints)

  __shared__ unsigned int As32[2][64][ASTR];   // [buf][m][kpair] : pair = (k,k+1) bf16
  __shared__ unsigned int Bs32[2][BR][BC];     // TRANSB: [n][kpair] ; else [kpair][n]

  const int tid  = threadIdx.x;
  const int z    = blockIdx.z;
  const long long aOff = (long long)(z / zdiv) * aOuter + (long long)(z % zdiv) * aInner;
  const long long bOff = (long long)(z / zdiv) * bOuter + (long long)(z % zdiv) * bInner;
  const long long cOff = (long long)(z / zdiv) * cOuter + (long long)(z % zdiv) * cInner;

  const int m0 = blockIdx.y * 64;
  const int n0 = blockIdx.x * BN;

  const int wave = tid >> 5;
  const int lane = tid & 31;
  const int mT   = wave & 3;
  const int nG   = (wave >> 2) * TPW;
  const int half = lane >> 4;
  const int l16  = lane & 15;

  // ---- staging helpers (global b128 loads -> packed bf16 LDS b64/b128 stores) ----
  auto stageBfloat = [&](int buf, int k0) {
    if constexpr (TRANSB) {                    // Bg[n][k] row-major -> [n][kpair]
#pragma unroll
      for (int e = 0; e < BN / 32; ++e) {
        int f = e * 256 + tid;
        int r = f >> 3, kq = f & 7;
        float4 v;
        if (!GUARD || (n0 + r) < N)
          v = *reinterpret_cast<const float4*>(
              Bm + bOff + (long long)(n0 + r) * ldb + (k0 + kq * 4));
        else
          v = float4{0.f, 0.f, 0.f, 0.f};
        uint2 p; p.x = pack2(v.x, v.y); p.y = pack2(v.z, v.w);
        *reinterpret_cast<uint2*>(&Bs32[buf][r][kq * 2]) = p;
      }
    } else {                                   // Bg[k][n] row-major -> [kpair][n]
#pragma unroll
      for (int e = 0; e < BN / 64; ++e) {
        int q = e * 256 + tid;
        int kp = q / (BN / 4), nq = q % (BN / 4);
        const float4* lo = reinterpret_cast<const float4*>(
            Bm + bOff + (long long)(k0 + 2 * kp) * ldb + n0);
        const float4* hi = reinterpret_cast<const float4*>(
            Bm + bOff + (long long)(k0 + 2 * kp + 1) * ldb + n0);
        float4 a4 = lo[nq], b4 = hi[nq];
        uint4 p;
        p.x = pack2(a4.x, b4.x); p.y = pack2(a4.y, b4.y);
        p.z = pack2(a4.z, b4.z); p.w = pack2(a4.w, b4.w);
        *reinterpret_cast<uint4*>(&Bs32[buf][kp][nq * 4]) = p;
      }
    }
  };

  auto stage = [&](int buf, int k0) {
    // A tile: 64x32 floats = 512 float4, 2 per thread
#pragma unroll
    for (int e = 0; e < 2; ++e) {
      int f = e * 256 + tid;
      int r = f >> 3, kq = f & 7;
      float4 v = *reinterpret_cast<const float4*>(
          A + aOff + (long long)(m0 + r) * lda + (k0 + kq * 4));
      uint2 p; p.x = pack2(v.x, v.y); p.y = pack2(v.z, v.w);
      *reinterpret_cast<uint2*>(&As32[buf][r][kq * 2]) = p;
    }
    if constexpr (TDMB) {
#if HAS_TDM
      if (tid < 32)                            // one wave issues the tile DMA
        tdm_load_tile_bf16(Bbf, bOff + (long long)n0 * ldb + k0, ldb,
                           N - n0, (unsigned)(uintptr_t)&Bs32[buf][0][0]);
#else
      stageBfloat(buf, k0);
#endif
    } else {
      stageBfloat(buf, k0);
    }
  };

  auto tdmWait = [&]() {
#if HAS_TDM
    if constexpr (TDMB) {
      if (tid < 32) __builtin_amdgcn_s_wait_tensorcnt((short)0);
    }
#endif
  };

  v8f acc[TPW];
#pragma unroll
  for (int t = 0; t < TPW; ++t) acc[t] = v8f{0.f,0.f,0.f,0.f,0.f,0.f,0.f,0.f};

  stage(0, 0);
  tdmWait();
  int p = 0;
  for (int k0 = 0; k0 < K; k0 += 32) {
    __syncthreads();                           // staging of buffer p complete
    int kn = k0 + 32;
    if (kn < K) {
      stage(p ^ 1, kn);                        // overlap next-tile staging with WMMAs
      // warm L2 two tiles ahead (global_prefetch_b8)
      if (kn + 32 < K) {
        __builtin_prefetch(A + aOff + (long long)(m0 + (tid >> 3)) * lda + (kn + 32), 0, 1);
        if constexpr (TRANSB && !TDMB)
          __builtin_prefetch(Bm + bOff + (long long)(n0 + (tid >> 3)) * ldb + (kn + 32), 0, 1);
        else if constexpr (!TRANSB)
          __builtin_prefetch(Bm + bOff + (long long)(kn + 32 + (tid >> 6)) * ldb + n0 + (tid & 63), 0, 1);
      }
    }

    // ---- fragments: contiguous k = 16*half + s (A and B alike), packed-pair dword reads ----
    Frag a;
#pragma unroll
    for (int v = 0; v < 8; ++v) {
      int kp = half * 8 + v;                          // half*8 + v (k = 16*half + s)
      a.u[v] = As32[p][mT * 16 + l16][kp];
    }
#pragma unroll
    for (int t = 0; t < TPW; ++t) {
      Frag b;
      if constexpr (TRANSB) {
#pragma unroll
        for (int v = 0; v < 8; ++v) b.u[v] = Bs32[p][(nG + t) * 16 + l16][half * 8 + v];
      } else {
#pragma unroll
        for (int v = 0; v < 8; ++v) b.u[v] = Bs32[p][half * 8 + v][(nG + t) * 16 + l16];
      }
      acc[t] = __builtin_amdgcn_wmma_f32_16x16x32_bf16(
          false, a.v, false, b.v, (short)0, acc[t], false, false);
    }
    if (kn < K) tdmWait();                     // DMA of buffer p^1 done before next barrier
    p ^= 1;
  }

  // ---- epilogue: C/D layout VGPR r -> M = r + 8*(lane/16), N = lane%16 ----
  const int cmBase = mT * 16 + 8 * half;
#pragma unroll
  for (int t = 0; t < TPW; ++t) {
    int gn = n0 + (nG + t) * 16 + l16;
    if (GUARD && gn >= N) continue;
#pragma unroll
    for (int r = 0; r < 8; ++r) {
      int gm = m0 + cmBase + r;
      float val = acc[t][r];
      if constexpr (SCORE) {
        val = val * scale - kPEN * dists[(long long)gm * N + gn];
        if (gn > gm) val = __uint_as_float(0xff800000u);   // -inf (causal)
      }
      if constexpr (BIAS) val += bias[gn];
      if constexpr (GELU) val = 0.5f * val * (1.0f + erff(val * 0.70710678118654752f));
      long long ci = cOff + (long long)gm * ldc + gn;
      if constexpr (RES) val += resid[ci];
      C[ci] = val;
    }
  }
}

// ---------------- elementwise / reduction kernels ----------------
__global__ __launch_bounds__(256) void tobf16_kernel(
    const float* __restrict__ src, unsigned short* __restrict__ dst)
{
  long long q = (long long)blockIdx.x * 256 + threadIdx.x;   // over N/4 quads
  float4 v = reinterpret_cast<const float4*>(src)[q];
  uint2 p; p.x = pack2(v.x, v.y); p.y = pack2(v.z, v.w);
  reinterpret_cast<uint2*>(dst)[q] = p;
}

__global__ __launch_bounds__(256) void embed_kernel(
    const int* __restrict__ idx, const float* __restrict__ emb,
    const float* __restrict__ pos, float* __restrict__ x)
{
  long long i = (long long)blockIdx.x * 256 + threadIdx.x;   // over B*T*D
  long long bt = i / kD;
  int d = (int)(i % kD);
  int t = (int)(bt % kT);
  x[i] = emb[(long long)idx[bt] * kD + d] + pos[(long long)t * kD + d];
}

__global__ __launch_bounds__(256) void ln_kernel(
    const float* __restrict__ x, const float* __restrict__ g,
    const float* __restrict__ b, float* __restrict__ out)
{
  __shared__ float red[256];
  const long long row = blockIdx.x;
  const float* xr = x + row * kD;
  float s = 0.f;
  for (int d = threadIdx.x; d < kD; d += 256) s += xr[d];
  red[threadIdx.x] = s; __syncthreads();
  for (int o = 128; o > 0; o >>= 1) { if (threadIdx.x < o) red[threadIdx.x] += red[threadIdx.x + o]; __syncthreads(); }
  float mean = red[0] / kD; __syncthreads();
  float v = 0.f;
  for (int d = threadIdx.x; d < kD; d += 256) { float t = xr[d] - mean; v += t * t; }
  red[threadIdx.x] = v; __syncthreads();
  for (int o = 128; o > 0; o >>= 1) { if (threadIdx.x < o) red[threadIdx.x] += red[threadIdx.x + o]; __syncthreads(); }
  float rstd = rsqrtf(red[0] / kD + 1e-5f);
  for (int d = threadIdx.x; d < kD; d += 256)
    out[row * kD + d] = (xr[d] - mean) * rstd * g[d] + b[d];
}

__global__ __launch_bounds__(256) void dists_kernel(
    const float* __restrict__ pos, float* __restrict__ dout, float* __restrict__ accum)
{
  __shared__ float red[256];
  long long idx = (long long)blockIdx.x * 256 + threadIdx.x;   // over T*T
  int i = (int)(idx / kT), j = (int)(idx % kT);
  float sq = 0.f;
#pragma unroll
  for (int c = 0; c < kDS; ++c) { float d = pos[i * kDS + c] - pos[j * kDS + c]; sq += d * d; }
  float dist = (sq > 0.f) ? sqrtf(sq) : 0.f;
  dout[idx] = dist;
  float tre = (i != j) ? 1.f / (dist + 1e-4f) : 0.f;
  red[threadIdx.x] = tre; __syncthreads();
  for (int o = 128; o > 0; o >>= 1) { if (threadIdx.x < o) red[threadIdx.x] += red[threadIdx.x + o]; __syncthreads(); }
  if (threadIdx.x == 0) atomicAdd(&accum[2], red[0]);
}

__global__ __launch_bounds__(256) void softmax_attn_kernel(
    float* __restrict__ w, const float* __restrict__ dists, float* __restrict__ accum)
{
  __shared__ float red[256];
  const long long row = blockIdx.x;                 // (b*H+h)*T + i
  const int i = (int)(row % kT);
  float* wr = w + row * kT;
  const float* dr = dists + (long long)i * kT;

  float m = -INFINITY;
  for (int j = threadIdx.x; j < kT; j += 256) m = fmaxf(m, wr[j]);
  red[threadIdx.x] = m; __syncthreads();
  for (int o = 128; o > 0; o >>= 1) { if (threadIdx.x < o) red[threadIdx.x] = fmaxf(red[threadIdx.x], red[threadIdx.x + o]); __syncthreads(); }
  m = red[0]; __syncthreads();

  float s = 0.f;
  for (int j = threadIdx.x; j < kT; j += 256) s += expf(wr[j] - m);
  red[threadIdx.x] = s; __syncthreads();
  for (int o = 128; o > 0; o >>= 1) { if (threadIdx.x < o) red[threadIdx.x] += red[threadIdx.x + o]; __syncthreads(); }
  float inv = 1.f / red[0]; __syncthreads();

  float tde = 0.f, tfe = 0.f;
  for (int j = threadIdx.x; j < kT; j += 256) {
    float p = expf(wr[j] - m) * inv;
    wr[j] = p;
    tde += p * dr[j];
    tfe -= p * logf(p + 1e-9f);
  }
  red[threadIdx.x] = tde; __syncthreads();
  for (int o = 128; o > 0; o >>= 1) { if (threadIdx.x < o) red[threadIdx.x] += red[threadIdx.x + o]; __syncthreads(); }
  if (threadIdx.x == 0) atomicAdd(&accum[0], red[0]);
  __syncthreads();
  red[threadIdx.x] = tfe; __syncthreads();
  for (int o = 128; o > 0; o >>= 1) { if (threadIdx.x < o) red[threadIdx.x] += red[threadIdx.x + o]; __syncthreads(); }
  if (threadIdx.x == 0) atomicAdd(&accum[1], red[0]);
}

__global__ __launch_bounds__(256) void ce_kernel(
    const float* __restrict__ logits, const int* __restrict__ targets, float* __restrict__ accum)
{
  __shared__ float red[256];
  const long long row = blockIdx.x;                 // over B*T
  const float* lr = logits + row * kV;

  float m = -INFINITY;
  for (int j = threadIdx.x; j < kV; j += 256) m = fmaxf(m, lr[j]);
  red[threadIdx.x] = m; __syncthreads();
  for (int o = 128; o > 0; o >>= 1) { if (threadIdx.x < o) red[threadIdx.x] = fmaxf(red[threadIdx.x], red[threadIdx.x + o]); __syncthreads(); }
  m = red[0]; __syncthreads();

  float s = 0.f;
  for (int j = threadIdx.x; j < kV; j += 256) s += expf(lr[j] - m);
  red[threadIdx.x] = s; __syncthreads();
  for (int o = 128; o > 0; o >>= 1) { if (threadIdx.x < o) red[threadIdx.x] += red[threadIdx.x + o]; __syncthreads(); }
  if (threadIdx.x == 0) {
    float lse = m + logf(red[0]);
    atomicAdd(&accum[3], -(lr[targets[row]] - lse));
  }
}

__global__ void finalize_kernel(const float* __restrict__ accum, float* __restrict__ out)
{
  float tde = accum[0] / (float)(kB * kH * kT);
  float tfe = accum[1] / (float)(kB * kH * kT);
  float tre = accum[2] / (float)((long long)kT * kT - kT);
  float ce  = accum[3] / (float)(kB * kT);
  out[0] = ce + kEW * tde + kFW * tfe + kRW * tre;  // loss
  out[1] = tde;
  out[2] = tfe;
  out[3] = tre;
}

// ---------------- host orchestration ----------------
extern "C" void kernel_launch(void* const* d_in, const int* in_sizes, int n_in,
                              void* d_out, int out_size, void* d_ws, size_t ws_size,
                              hipStream_t stream) {
  (void)in_sizes; (void)n_in; (void)out_size; (void)ws_size;
  const int*   idx     = (const int*)  d_in[0];
  const int*   targets = (const int*)  d_in[1];
  const float* emb     = (const float*)d_in[2];
  const float* pos_emb = (const float*)d_in[3];
  const float* positions = (const float*)d_in[4];
  const float* ln1_g = (const float*)d_in[5];
  const float* ln1_b = (const float*)d_in[6];
  const float* ln2_g = (const float*)d_in[7];
  const float* ln2_b = (const float*)d_in[8];
  const float* lnf_g = (const float*)d_in[9];
  const float* lnf_b = (const float*)d_in[10];
  const float* qkv_w = (const float*)d_in[11];
  const float* proj_w = (const float*)d_in[12];
  const float* w1 = (const float*)d_in[13];
  const float* b1 = (const float*)d_in[14];
  const float* w2 = (const float*)d_in[15];
  const float* b2 = (const float*)d_in[16];

  float* outf  = (float*)d_out;
  float* ws    = (float*)d_ws;
  float* accum = ws + WS_ACC;
  float* x     = ws + WS_X;
  float* hbuf  = ws + WS_H;
  float* qkv   = ws + WS_QKV;
  float* att   = ws + WS_ATT;
  float* mlp   = ws + WS_MLP;
  unsigned short* embbf = (unsigned short*)(ws + WS_EMBBF);

  (void)hipMemsetAsync(accum, 0, 4 * sizeof(float), stream);

  const dim3 blk(256);
  const int BT = kB * kT;                          // 2048

  // emb -> bf16 (for TDM-fed lm-head GEMM); V*D = 51,463,168 = 50257 * 1024
  tobf16_kernel<<<dim3((unsigned)((long long)kV * kD / 1024)), blk, 0, stream>>>(emb, embbf);

  // x = emb[idx] + pos_emb
  embed_kernel<<<dim3((unsigned)((long long)BT * kD / 256)), blk, 0, stream>>>(idx, emb, pos_emb, x);

  const float scale = 1.0f / sqrtf((float)kDH);

  for (int l = 0; l < kL; ++l) {
    float* wl = outf + W_OFF + (long long)l * W_N;
    float* dl = outf + DT_OFF + (long long)l * D_N;

    dists_kernel<<<dim3((unsigned)((long long)kT * kT / 256)), blk, 0, stream>>>(
        positions + (long long)l * kT * kDS, dl, accum);

    ln_kernel<<<dim3(BT), blk, 0, stream>>>(x, ln1_g + l * kD, ln1_b + l * kD, hbuf);

    // qkv = h @ qkv_w[l] : [2048,1024]x[1024,3072]
    gemm_wmma<128,false,false,false,false,false,false,false>
        <<<dim3(3 * kD / 128, BT / 64, 1), blk, 0, stream>>>(
        hbuf, qkv_w + (long long)l * kD * 3 * kD, nullptr, nullptr, nullptr, qkv, nullptr,
        BT, 3 * kD, kD, kD, 3 * kD, 3 * kD, 0,0,0,0,0,0, 1, 1.f);

    // scores: per (b,h): q@k^T*scale - PEN*dists, causal -> all_w[l]
    gemm_wmma<128,true,false,false,false,true,false,false>
        <<<dim3(kT / 128, kT / 64, kB * kH), blk, 0, stream>>>(
        qkv /*q*/, qkv + kD /*k*/, nullptr, nullptr, nullptr, wl, dl,
        kT, kT, kDH, 3 * kD, 3 * kD, kT,
        (long long)kT * 3 * kD, kDH,                       // A batch strides (b,h)
        (long long)kT * 3 * kD, kDH,                       // B batch strides
        (long long)kH * kT * kT, (long long)kT * kT,       // C batch strides
        kH, scale);

    softmax_attn_kernel<<<dim3(kB * kH * kT), blk, 0, stream>>>(wl, dl, accum);

    // att = w @ v : per (b,h): [1024,1024]x[1024,64]
    gemm_wmma<64,false,false,false,false,false,false,false>
        <<<dim3(1, kT / 64, kB * kH), blk, 0, stream>>>(
        wl, qkv + 2 * kD /*v*/, nullptr, nullptr, nullptr, att, nullptr,
        kT, kDH, kT, kT, 3 * kD, kD,
        (long long)kH * kT * kT, (long long)kT * kT,
        (long long)kT * 3 * kD, kDH,
        (long long)kT * kD, kDH,
        kH, 1.f);

    // x += att @ proj_w[l]
    gemm_wmma<128,false,false,false,true,false,false,false>
        <<<dim3(kD / 128, BT / 64, 1), blk, 0, stream>>>(
        att, proj_w + (long long)l * kD * kD, nullptr, nullptr, x, x, nullptr,
        BT, kD, kD, kD, kD, kD, 0,0,0,0,0,0, 1, 1.f);

    ln_kernel<<<dim3(BT), blk, 0, stream>>>(x, ln2_g + l * kD, ln2_b + l * kD, hbuf);

    // mlp = gelu(h2 @ w1 + b1)
    gemm_wmma<128,false,true,true,false,false,false,false>
        <<<dim3(kDF / 128, BT / 64, 1), blk, 0, stream>>>(
        hbuf, w1 + (long long)l * kD * kDF, nullptr, b1 + (long long)l * kDF, nullptr, mlp, nullptr,
        BT, kDF, kD, kD, kDF, kDF, 0,0,0,0,0,0, 1, 1.f);

    // x += mlp @ w2 + b2
    gemm_wmma<128,false,true,false,true,false,false,false>
        <<<dim3(kD / 128, BT / 64, 1), blk, 0, stream>>>(
        mlp, w2 + (long long)l * kDF * kD, nullptr, b2 + (long long)l * kD, x, x, nullptr,
        BT, kD, kDF, kDF, kD, kD, 0,0,0,0,0,0, 1, 1.f);
  }

  // final LN, tied lm-head: logits = ln_f(x) @ emb^T
  // (emb bf16 [V,D] -> TRANSB, B tiles via Tensor Data Mover, ragged N -> GUARD)
  ln_kernel<<<dim3(BT), blk, 0, stream>>>(x, lnf_g, lnf_b, hbuf);
  gemm_wmma<128,true,false,false,false,false,true,true>
      <<<dim3((kV + 127) / 128, BT / 64, 1), blk, 0, stream>>>(
      hbuf, emb, embbf, nullptr, nullptr, outf, nullptr,
      BT, kV, kD, kD, kD, kV, 0,0,0,0,0,0, 1, 1.f);

  ce_kernel<<<dim3(BT), blk, 0, stream>>>(outf, targets, accum);
  finalize_kernel<<<dim3(1), dim3(1), 0, stream>>>(accum, outf + SCALAR_OFF);
}
